// MoBAAttention_79087527788592
// MI455X (gfx1250) — matrix operand order repaired and checked by reading, not compiled, not run on hardware
//
#include <hip/hip_runtime.h>

// ---------------- problem constants ----------------
#define Hh   16      // heads
#define Sq   2048    // sequence
#define Dm   2048    // model dim
#define DHd  128     // head dim
#define BSZ  256     // MoBA block size
#define NB   8       // number of key blocks
#define KVPAD  136   // padded LDS row (f16) for K chunk rows (64x128)
#define VTPAD  72    // padded LDS row (f16) for V^T chunk rows (128x64)
#define PBPAD  72    // padded LDS row (f16) for P (32x64 per wave)

typedef __attribute__((ext_vector_type(16))) _Float16 v16h;
typedef __attribute__((ext_vector_type(8)))  _Float16 v8h;
typedef __attribute__((ext_vector_type(8)))  float    v8f;
typedef __attribute__((ext_vector_type(4)))  int      i4;

// ---------------- CDNA5 async global->LDS path (guarded) ----------------
#if defined(__gfx1250__) && __has_builtin(__builtin_amdgcn_global_load_async_to_lds_b128)
#define HAS_ASYNC_LDS 1
#else
#define HAS_ASYNC_LDS 0
#endif

#if HAS_ASYNC_LDS
typedef __attribute__((address_space(1))) i4 gi4;   // global int4
typedef __attribute__((address_space(3))) i4 li4;   // LDS int4
#endif

static __device__ __forceinline__ void async_copy16(void* lds, const void* gl) {
#if HAS_ASYNC_LDS
  __builtin_amdgcn_global_load_async_to_lds_b128((gi4*)gl, (li4*)lds, 0, 0);
#else
  *(v8h*)lds = *(const v8h*)gl;   // synchronous fallback
#endif
}

#if HAS_ASYNC_LDS && __has_builtin(__builtin_amdgcn_s_wait_asynccnt)
#define ASYNC_WAIT(n) __builtin_amdgcn_s_wait_asynccnt(n)
#else
#define ASYNC_WAIT(n) do {} while (0)
#endif

static __device__ __forceinline__ v16h pack16(v8h lo, v8h hi) {
  v16h r;
#pragma unroll
  for (int i = 0; i < 8; ++i) { r[i] = lo[i]; r[8 + i] = hi[i]; }
  return r;
}

// ---------------- f32 -> f16 convert ----------------
__global__ void k_cvt(const float* __restrict__ in, _Float16* __restrict__ out, int n) {
  int i = blockIdx.x * blockDim.x + threadIdx.x;
  int stride = gridDim.x * blockDim.x;
  for (; i < n; i += stride) out[i] = (_Float16)in[i];
}

// ---------------- NT GEMM: C[M,N] = A[M,K] * B[N,K]^T + bias ----------------
// mode 0: f16 head-major      Chead[col>>7][row][col&127]  (Q,K projections)
// mode 2: f16 head-transposed Chead[col>>7][col&127][row]  (V projection)
// mode 1: f32 row-major       Cflat[row*Dm + col]          (output projection)
__global__ __launch_bounds__(256) void k_gemm_nt(
    const _Float16* __restrict__ A, const _Float16* __restrict__ Bm,
    const float* __restrict__ bias,
    _Float16* __restrict__ Chead, float* __restrict__ Cflat, int mode) {
  __shared__ _Float16 As[2][128][40];
  __shared__ _Float16 Bs[2][128][40];

  const int tid = threadIdx.x;
  const int wave = tid >> 5, lane = tid & 31;
  const int l = lane & 15, hi = lane >> 4;
  const int wr = wave >> 1, wc = wave & 1;      // 4x2 wave grid -> 32x64 per wave
  const int m0 = blockIdx.x * 128, n0 = blockIdx.y * 128;

  v8f acc[2][4];
#pragma unroll
  for (int t = 0; t < 2; ++t)
#pragma unroll
    for (int ct = 0; ct < 4; ++ct)
#pragma unroll
      for (int r = 0; r < 8; ++r) acc[t][ct][r] = 0.f;

  const int lrow = tid >> 1;            // 0..127
  const int lseg = (tid & 1) * 16;      // 0 or 16

  auto issue_tile = [&](int k0, int b) {
    const size_t ga = (size_t)(m0 + lrow) * Dm + k0 + lseg;
    const size_t gb = (size_t)(n0 + lrow) * Dm + k0 + lseg;
    async_copy16(&As[b][lrow][lseg],     &A[ga]);
    async_copy16(&As[b][lrow][lseg + 8], &A[ga + 8]);
    async_copy16(&Bs[b][lrow][lseg],     &Bm[gb]);
    async_copy16(&Bs[b][lrow][lseg + 8], &Bm[gb + 8]);
  };

  const int NT = Dm / 32;
  issue_tile(0, 0);
  for (int kt = 0; kt < NT; ++kt) {
    const int b = kt & 1;
    if (kt + 1 < NT) { issue_tile((kt + 1) * 32, b ^ 1); ASYNC_WAIT(4); }
    else             { ASYNC_WAIT(0); }
    __syncthreads();

    v16h af[2], bf[4];
#pragma unroll
    for (int t = 0; t < 2; ++t) {
      const _Float16* ar = &As[b][wr * 32 + t * 16 + l][0];
      af[t] = pack16(*(const v8h*)(ar + hi * 8), *(const v8h*)(ar + hi * 8 + 16));
    }
#pragma unroll
    for (int ct = 0; ct < 4; ++ct) {
      const _Float16* br = &Bs[b][wc * 64 + ct * 16 + l][0];
      bf[ct] = pack16(*(const v8h*)(br + hi * 8), *(const v8h*)(br + hi * 8 + 16));
    }
#pragma unroll
    for (int ct = 0; ct < 4; ++ct)
#pragma unroll
      for (int t = 0; t < 2; ++t)
        acc[t][ct] = __builtin_amdgcn_wmma_f32_16x16x32_f16(
            false, af[t], false, bf[ct], (short)0, acc[t][ct], false, false);
    __syncthreads();
  }

#pragma unroll
  for (int t = 0; t < 2; ++t)
#pragma unroll
    for (int ct = 0; ct < 4; ++ct) {
      const int col = n0 + wc * 64 + ct * 16 + l;
      const float bv = bias[col];
#pragma unroll
      for (int r = 0; r < 8; ++r) {
        const int row = m0 + wr * 32 + t * 16 + r + hi * 8;
        const float v = acc[t][ct][r] + bv;
        if (mode == 0)
          Chead[(((size_t)(col >> 7)) * Sq + row) * DHd + (col & 127)] = (_Float16)v;
        else if (mode == 2)
          Chead[(((size_t)(col >> 7)) * DHd + (col & 127)) * Sq + row] = (_Float16)v;
        else
          Cflat[(size_t)row * Dm + col] = v;
      }
    }
}

// ---------------- per-block mean key ----------------
__global__ __launch_bounds__(128) void k_kmean(const _Float16* __restrict__ Kh,
                                               float* __restrict__ km) {
  const int b = blockIdx.x;          // h*NB + nb
  const int h = b >> 3, nb = b & 7;
  const int d = threadIdx.x;
  float s = 0.f;
  for (int t = 0; t < BSZ; ++t)
    s += (float)Kh[((size_t)h * Sq + nb * BSZ + t) * DHd + d];
  km[(size_t)b * DHd + d] = s * (1.f / BSZ);
}

// ---------------- MoBA routing: top-3 strictly-past blocks ----------------
__global__ __launch_bounds__(256) void k_route(const _Float16* __restrict__ Qh,
                                               const float* __restrict__ km,
                                               unsigned* __restrict__ sel) {
  const int wave = threadIdx.x >> 5, lane = threadIdx.x & 31;
  const int g = blockIdx.x * 8 + wave;  // h*Sq + s
  const int h = g >> 11, s = g & (Sq - 1);

  const _Float16* qp = Qh + ((size_t)h * Sq + s) * DHd;
  float qv[4];
#pragma unroll
  for (int i = 0; i < 4; ++i) qv[i] = (float)qp[lane * 4 + i];

  float route[NB];
#pragma unroll
  for (int nb = 0; nb < NB; ++nb) {
    const float* kp = km + (size_t)(h * NB + nb) * DHd;
    float p = 0.f;
#pragma unroll
    for (int i = 0; i < 4; ++i) p += qv[i] * kp[lane * 4 + i];
#pragma unroll
    for (int off = 16; off; off >>= 1) p += __shfl_xor(p, off, 32);
    route[nb] = p;
  }
  const int qblk = s >> 8;
#pragma unroll
  for (int nb = 0; nb < NB; ++nb)
    if (nb >= qblk) route[nb] = -3.0e38f;     // mask current + future blocks

  unsigned mask = 0;
#pragma unroll
  for (int t = 0; t < 3; ++t) {               // TOPK = 3
    float best = -3.0e38f; int bi = -1;
#pragma unroll
    for (int nb = 0; nb < NB; ++nb)
      if (route[nb] > best) { best = route[nb]; bi = nb; }
    if (bi >= 0) { mask |= 1u << bi; route[bi] = -3.2e38f; }
  }
  if (lane == 0) sel[g] = mask;
}

// ---------------- streaming WMMA attention ----------------
// mode 0: MoBA pass over past blocks with routing mask, attn = result
// mode 1: self pass (own block, causal), attn += result
// K is head-major [H][S][Dh]; V is head-transposed [H][Dh][S].
__global__ __launch_bounds__(256) void k_attn(
    const _Float16* __restrict__ Qh, const _Float16* __restrict__ Kh,
    const _Float16* __restrict__ Vt, const unsigned* __restrict__ sel,
    float* __restrict__ attn, int mode) {
  const int n = blockIdx.x, h = blockIdx.y;
  const int tid = threadIdx.x, wave = tid >> 5, lane = tid & 31;
  const int l = lane & 15, hi = lane >> 4;

  if (mode == 0 && n == 0) {  // no past blocks -> moba output is zero
    float* dst = attn + (size_t)h * Sq * DHd;
    for (int i = tid; i < BSZ * DHd; i += 256) dst[i] = 0.f;
    return;
  }

  extern __shared__ char smem[];
  // buffer base helpers (no pointer arrays -> no static addrspacecast initializers)
  auto KbP = [&](int b) -> _Float16 (*)[KVPAD] {
    return (_Float16(*)[KVPAD])(smem + b * 17408);
  };
  auto VbP = [&](int b) -> _Float16 (*)[VTPAD] {
    return (_Float16(*)[VTPAD])(smem + 34816 + b * 18432);
  };
  _Float16 (*Pbuf)[PBPAD] = (_Float16(*)[PBPAD])(smem + 71680) + wave * 32;

  const float scale = 0.088388347648318447f;  // 1/sqrt(128)

  // resident Q fragments: 2 row tiles x 4 k-chunks
  v16h aq[2][4];
#pragma unroll
  for (int t = 0; t < 2; ++t) {
    const _Float16* qr =
        Qh + ((size_t)h * Sq + n * BSZ + wave * 32 + t * 16 + l) * DHd;
#pragma unroll
    for (int kc = 0; kc < 4; ++kc)
      aq[t][kc] = pack16(*(const v8h*)(qr + kc * 32 + hi * 8),
                         *(const v8h*)(qr + kc * 32 + hi * 8 + 16));
  }

  unsigned selr[2][8];
#pragma unroll
  for (int t = 0; t < 2; ++t)
#pragma unroll
    for (int r = 0; r < 8; ++r)
      selr[t][r] = sel[(size_t)h * Sq + n * BSZ + wave * 32 + t * 16 + r + hi * 8];

  float ms[2][8], ls[2][8];
  v8f acc[2][8];
#pragma unroll
  for (int t = 0; t < 2; ++t)
#pragma unroll
    for (int r = 0; r < 8; ++r) { ms[t][r] = -1e30f; ls[t][r] = 0.f; }
#pragma unroll
  for (int t = 0; t < 2; ++t)
#pragma unroll
    for (int vt = 0; vt < 8; ++vt)
#pragma unroll
      for (int r = 0; r < 8; ++r) acc[t][vt][r] = 0.f;

  const int jbeg = mode ? n : 0;
  const int NC = (mode ? 1 : n) * 4;   // 64-key chunks to process

  // stage chunk ci (j = jbeg + ci/4, c = ci%4) into buffer b
  auto issue_chunk = [&](int ci, int b) {
    const int j = jbeg + (ci >> 2), c = ci & 3;
    _Float16 (*Kw)[KVPAD] = KbP(b);
    _Float16 (*Vw)[VTPAD] = VbP(b);
#pragma unroll
    for (int p = 0; p < 4; ++p) {                       // K: 64 rows x 128 f16
      const int idx = p * 256 + tid;
      const int row = idx >> 4, c16 = (idx & 15) * 8;
      async_copy16(&Kw[row][c16],
                   &Kh[((size_t)h * Sq + j * BSZ + c * 64 + row) * DHd + c16]);
    }
#pragma unroll
    for (int p = 0; p < 4; ++p) {                       // V^T: 128 rows x 64 f16
      const int idx = p * 256 + tid;
      const int row = idx >> 3, c8 = (idx & 7) * 8;
      async_copy16(&Vw[row][c8],
                   &Vt[((size_t)h * DHd + row) * Sq + j * BSZ + c * 64 + c8]);
    }
  };

  issue_chunk(0, 0);
  for (int ci = 0; ci < NC; ++ci) {
    const int cur = ci & 1;
    const int j = jbeg + (ci >> 2), c = ci & 3;
    if (ci + 1 < NC) { issue_chunk(ci + 1, cur ^ 1); ASYNC_WAIT(8); }
    else             { ASYNC_WAIT(0); }
    __syncthreads();

    _Float16 (*Kc)[KVPAD] = KbP(cur);
    _Float16 (*Vc)[VTPAD] = VbP(cur);

    // ---- scores: S = Q * K^T (32 rows x 64 keys per wave) ----
    v8f sc[2][4];
#pragma unroll
    for (int t = 0; t < 2; ++t)
#pragma unroll
      for (int ct = 0; ct < 4; ++ct)
#pragma unroll
        for (int r = 0; r < 8; ++r) sc[t][ct][r] = 0.f;

#pragma unroll
    for (int ct = 0; ct < 4; ++ct) {
      const _Float16* kr = &Kc[ct * 16 + l][0];
#pragma unroll
      for (int kc = 0; kc < 4; ++kc) {
        v16h bk = pack16(*(const v8h*)(kr + kc * 32 + hi * 8),
                         *(const v8h*)(kr + kc * 32 + hi * 8 + 16));
#pragma unroll
        for (int t = 0; t < 2; ++t)
          sc[t][ct] = __builtin_amdgcn_wmma_f32_16x16x32_f16(
              false, aq[t][kc], false, bk, (short)0, sc[t][ct], false, false);
      }
    }

    // ---- mask + streaming softmax update ----
#pragma unroll
    for (int t = 0; t < 2; ++t) {
#pragma unroll
      for (int r = 0; r < 8; ++r) {
        const int rl = wave * 32 + t * 16 + r + hi * 8;  // local query row
        float mr = -3.0e38f;
#pragma unroll
        for (int ct = 0; ct < 4; ++ct) {
          float v = sc[t][ct][r] * scale;
          bool keep;
          if (mode) {
            const int kg = c * 64 + ct * 16 + l;         // key index in block
            keep = (kg <= rl);                           // causal
          } else {
            keep = ((selr[t][r] >> j) & 1u) != 0u;       // routed block
          }
          v = keep ? v : -1e30f;
          sc[t][ct][r] = v;
          mr = fmaxf(mr, v);
        }
#pragma unroll
        for (int off = 1; off < 16; off <<= 1)
          mr = fmaxf(mr, __shfl_xor(mr, off, 16));
        const float mnew = fmaxf(ms[t][r], mr);
        const float resc = __expf(ms[t][r] - mnew);
        ms[t][r] = mnew;

        float rs = 0.f;
#pragma unroll
        for (int ct = 0; ct < 4; ++ct) {
          const float p = __expf(sc[t][ct][r] - mnew);
          rs += p;
          Pbuf[t * 16 + r + hi * 8][ct * 16 + l] = (_Float16)p;
        }
#pragma unroll
        for (int off = 1; off < 16; off <<= 1)
          rs += __shfl_xor(rs, off, 16);
        ls[t][r] = ls[t][r] * resc + rs;
#pragma unroll
        for (int vt = 0; vt < 8; ++vt) acc[t][vt][r] *= resc;
      }
    }

    // ---- acc += P * V  (V^T rows are dh, contiguous keys) ----
    v16h ap[2][2];
#pragma unroll
    for (int t = 0; t < 2; ++t) {
      const _Float16* pr = &Pbuf[t * 16 + l][0];
#pragma unroll
      for (int kc2 = 0; kc2 < 2; ++kc2)
        ap[t][kc2] = pack16(*(const v8h*)(pr + kc2 * 32 + hi * 8),
                            *(const v8h*)(pr + kc2 * 32 + hi * 8 + 16));
    }
#pragma unroll
    for (int vt = 0; vt < 8; ++vt) {
      const _Float16* vr = &Vc[vt * 16 + l][0];
#pragma unroll
      for (int kc2 = 0; kc2 < 2; ++kc2) {
        v16h bvf = pack16(*(const v8h*)(vr + kc2 * 32 + hi * 8),
                          *(const v8h*)(vr + kc2 * 32 + hi * 8 + 16));
#pragma unroll
        for (int t = 0; t < 2; ++t)
          acc[t][vt] = __builtin_amdgcn_wmma_f32_16x16x32_f16(
              false, ap[t][kc2], false, bvf, (short)0, acc[t][vt], false, false);
      }
    }
    __syncthreads();
  }

  // ---- finalize: divide by row sums, write / accumulate ----
#pragma unroll
  for (int t = 0; t < 2; ++t) {
    float inv[8];
#pragma unroll
    for (int r = 0; r < 8; ++r) inv[r] = (ls[t][r] > 0.f) ? 1.f / ls[t][r] : 0.f;
#pragma unroll
    for (int vt = 0; vt < 8; ++vt)
#pragma unroll
      for (int r = 0; r < 8; ++r) {
        const size_t o =
            ((size_t)h * Sq + n * BSZ + wave * 32 + t * 16 + r + hi * 8) * DHd +
            vt * 16 + l;
        const float v = acc[t][vt][r] * inv[r];
        if (mode) attn[o] += v; else attn[o] = v;
      }
  }
}

// ---------------- [H][S][Dh] f32 -> [S][D] f16 ----------------
__global__ void k_attn_to_f16(const float* __restrict__ attn,
                              _Float16* __restrict__ out) {
  int i = blockIdx.x * blockDim.x + threadIdx.x;
  const int total = Hh * Sq * DHd;
  const int stride = gridDim.x * blockDim.x;
  for (; i < total; i += stride) {
    const int h = i / (Sq * DHd);
    const int rem = i - h * (Sq * DHd);
    const int s = rem / DHd;
    const int d = rem - s * DHd;
    out[(size_t)s * Dm + h * DHd + d] = (_Float16)attn[i];
  }
}

// ---------------- host launch ----------------
extern "C" void kernel_launch(void* const* d_in, const int* in_sizes, int n_in,
                              void* d_out, int out_size, void* d_ws, size_t ws_size,
                              hipStream_t stream) {
  (void)in_sizes; (void)n_in; (void)out_size; (void)ws_size;
  const float* query = (const float*)d_in[0];
  const float* key   = (const float*)d_in[1];
  const float* value = (const float*)d_in[2];
  const float* Wq = (const float*)d_in[3];
  const float* bq = (const float*)d_in[4];
  const float* Wk = (const float*)d_in[5];
  const float* bk = (const float*)d_in[6];
  const float* Wv = (const float*)d_in[7];
  const float* bv = (const float*)d_in[8];
  const float* Wo = (const float*)d_in[9];
  const float* bo = (const float*)d_in[10];
  float* out = (float*)d_out;

  char* base = (char*)d_ws;
  size_t off = 0;
  auto take = [&](size_t bytes) -> void* {
    void* q = base + off;
    off = (off + bytes + 255) & ~(size_t)255;
    return q;
  };
  const size_t SD = (size_t)Sq * Dm;
  _Float16* xq16 = (_Float16*)take(SD * 2);
  _Float16* xk16 = (_Float16*)take(SD * 2);
  _Float16* xv16 = (_Float16*)take(SD * 2);
  _Float16* wq16 = (_Float16*)take(SD * 2);
  _Float16* wk16 = (_Float16*)take(SD * 2);
  _Float16* wv16 = (_Float16*)take(SD * 2);
  _Float16* wo16 = (_Float16*)take(SD * 2);
  _Float16* Qhb  = (_Float16*)take((size_t)Hh * Sq * DHd * 2);
  _Float16* Khb  = (_Float16*)take((size_t)Hh * Sq * DHd * 2);
  _Float16* Vtb  = (_Float16*)take((size_t)Hh * Sq * DHd * 2);  // [H][Dh][S]
  float*    km   = (float*)take((size_t)Hh * NB * DHd * 4);
  unsigned* selm = (unsigned*)take((size_t)Hh * Sq * 4);
  float*    attn = (float*)take((size_t)Hh * Sq * DHd * 4);
  _Float16* attT = (_Float16*)take(SD * 2);

  // 1) convert activations + weights to f16
  k_cvt<<<2048, 256, 0, stream>>>(query, xq16, (int)SD);
  k_cvt<<<2048, 256, 0, stream>>>(key,   xk16, (int)SD);
  k_cvt<<<2048, 256, 0, stream>>>(value, xv16, (int)SD);
  k_cvt<<<2048, 256, 0, stream>>>(Wq, wq16, (int)SD);
  k_cvt<<<2048, 256, 0, stream>>>(Wk, wk16, (int)SD);
  k_cvt<<<2048, 256, 0, stream>>>(Wv, wv16, (int)SD);
  k_cvt<<<2048, 256, 0, stream>>>(Wo, wo16, (int)SD);

  // 2) QKV projections (V written transposed per head)
  dim3 gg(Dm / 128, Dm / 128);
  k_gemm_nt<<<gg, 256, 0, stream>>>(xq16, wq16, bq, Qhb, nullptr, 0);
  k_gemm_nt<<<gg, 256, 0, stream>>>(xk16, wk16, bk, Khb, nullptr, 0);
  k_gemm_nt<<<gg, 256, 0, stream>>>(xv16, wv16, bv, Vtb, nullptr, 2);

  // 3) routing
  k_kmean<<<Hh * NB, 128, 0, stream>>>(Khb, km);
  k_route<<<(Hh * Sq) / 8, 256, 0, stream>>>(Qhb, km, selm);

  // 4) attention (moba pass then self pass, stream-ordered accumulation)
  const size_t smem = 71680 + (size_t)8 * 32 * PBPAD * 2;   // 108,544 B
  (void)hipFuncSetAttribute(reinterpret_cast<const void*>(k_attn),
                            hipFuncAttributeMaxDynamicSharedMemorySize, (int)smem);
  k_attn<<<dim3(NB, Hh), 256, smem, stream>>>(Qhb, Khb, Vtb, selm, attn, 0);
  k_attn<<<dim3(NB, Hh), 256, smem, stream>>>(Qhb, Khb, Vtb, selm, attn, 1);

  // 5) output projection
  k_attn_to_f16<<<2048, 256, 0, stream>>>(attn, attT);
  k_gemm_nt<<<gg, 256, 0, stream>>>(attT, wo16, bo, nullptr, out, 1);
}
